// Derivative_78898549227959
// MI455X (gfx1250) — matrix operand order repaired and checked by reading, not compiled, not run on hardware
//
#include <hip/hip_runtime.h>
#include <hip/hip_bf16.h>

// Problem dims (fixed by the reference)
#define B_SZ 8192
#define D_SZ 129     // nodes / input width
#define H_SZ 64      // hidden
#define KPAD 160     // D padded to multiple of 32 for wmma f16 K=32
#define NKT1 5       // KPAD / 32
#define NKT2 2       // H   / 32
#define NNT  4       // H   / 16

typedef __attribute__((ext_vector_type(16))) _Float16 v16h;
typedef __attribute__((ext_vector_type(8)))  _Float16 v8h;
typedef __attribute__((ext_vector_type(8)))  float    v8f;

// ---------------------------------------------------------------------------
// Precompute 1: straight-through forward on even columns, f16 convert, K-pad.
// xpad[b][k] row-major, k in [0,160), zeros in padding.
// ---------------------------------------------------------------------------
__global__ __launch_bounds__(256) void prep_x_kernel(const float* __restrict__ inp,
                                                     const int* __restrict__ dmask,
                                                     _Float16* __restrict__ xpad) {
  int idx = blockIdx.x * 256 + threadIdx.x;        // over B*KPAD (exact multiple)
  int b = idx / KPAD;
  int k = idx - b * KPAD;
  float v = 0.f;
  if (k < D_SZ) {
    float t = inp[b * D_SZ + k];
    v = dmask[k] ? (t > 0.f ? 1.f : 0.f) : t;      // straight-through fwd == hard
  }
  xpad[idx] = (_Float16)v;
}

// ---------------------------------------------------------------------------
// Precompute 2: fold adjacency into W1, convert f16, swizzle into WMMA
// B-fragment order:  frag[n][kt][nt][lane][j]  (j == K offset within half)
//   element = A[n][d] * W1[n][d][h],  d = kt*32 + (j/8)*16 + (lane/16)*8 + j%8,
//                                     h = nt*16 + lane%16;  d >= 129 -> 0.
// ---------------------------------------------------------------------------
__global__ __launch_bounds__(256) void fold_w1_kernel(const float* __restrict__ W1,
                                                      const float* __restrict__ A,
                                                      _Float16* __restrict__ w1f) {
  int idx = blockIdx.x * 256 + threadIdx.x;        // over 129*5*4*32*16 (exact)
  int j    = idx & 15;
  int t    = idx >> 4;
  int lane = t & 31;  t >>= 5;
  int nt   = t & 3;   t >>= 2;
  int kt   = t % NKT1;
  int n    = t / NKT1;
  int d = kt * 32 + (j >> 3) * 16 + (lane >> 4) * 8 + (j & 7);
  int h = nt * 16 + (lane & 15);
  float v = 0.f;
  if (d < D_SZ)
    v = A[n * D_SZ + d] * W1[((size_t)n * D_SZ + d) * H_SZ + h];
  w1f[idx] = (_Float16)v;
}

// Same swizzle for W2 (K = 64 exactly, no padding needed).
__global__ __launch_bounds__(256) void fold_w2_kernel(const float* __restrict__ W2,
                                                      _Float16* __restrict__ w2f) {
  int idx = blockIdx.x * 256 + threadIdx.x;        // over 129*2*4*32*16 (exact)
  int j    = idx & 15;
  int t    = idx >> 4;
  int lane = t & 31;  t >>= 5;
  int nt   = t & 3;   t >>= 2;
  int kt   = t % NKT2;
  int n    = t / NKT2;
  int k = kt * 32 + (j >> 3) * 16 + (lane >> 4) * 8 + (j & 7);          // hidden-in (K), always < 64
  int h = nt * 16 + (lane & 15);                   // hidden-out (N)
  w2f[idx] = (_Float16)W2[((size_t)n * H_SZ + k) * H_SZ + h];
}

// ---------------------------------------------------------------------------
// Fused per-node MLP. Grid: (B/128, D). Block: 256 threads = 8 waves.
// Wave w handles batch rows [bx*128 + w*16, +16) for node n = blockIdx.y.
// ---------------------------------------------------------------------------
__global__ __launch_bounds__(256) void
fused_node_mlp_kernel(const _Float16* __restrict__ xpad,
                      const _Float16* __restrict__ w1f,
                      const _Float16* __restrict__ w2f,
                      const float* __restrict__ b1,
                      const float* __restrict__ b2,
                      const float* __restrict__ W3,
                      const float* __restrict__ b3,
                      float* __restrict__ out) {
  __shared__ _Float16 sW1[NKT1 * NNT * 512];   // 20 KB: W1 frags for node n
  __shared__ _Float16 sW2[NKT2 * NNT * 512];   //  8 KB: W2 frags for node n
  __shared__ _Float16 sH1[8 * 16 * H_SZ];      // 16 KB: per-wave h1 restage

  const int n   = blockIdx.y;
  const int tid = threadIdx.x;

  // ---- stage this node's pre-swizzled weights into LDS (b128 copies) ----
  {
    const uint4* g1 = (const uint4*)(w1f + (size_t)n * (NKT1 * NNT * 512));
    uint4* s1 = (uint4*)sW1;
#pragma unroll
    for (int i = 0; i < 5; ++i) s1[tid + i * 256] = g1[tid + i * 256];   // 1280 x 16B
    const uint4* g2 = (const uint4*)(w2f + (size_t)n * (NKT2 * NNT * 512));
    uint4* s2 = (uint4*)sW2;
#pragma unroll
    for (int i = 0; i < 2; ++i) s2[tid + i * 256] = g2[tid + i * 256];   // 512 x 16B
  }
  __syncthreads();

  const int wave  = tid >> 5;
  const int lane  = tid & 31;
  const int khalf = lane >> 4;    // which K-half this lane holds in A/B frags
  const int lrow  = lane & 15;    // A: row M; B/C: column N
  const int r0    = blockIdx.x * 128 + wave * 16;

  // ---- A fragments of x (row-major global, two b128 per fragment) ----
  v16h a[NKT1];
  const _Float16* xr = xpad + (size_t)(r0 + lrow) * KPAD + khalf * 8;
#pragma unroll
  for (int kt = 0; kt < NKT1; ++kt) {
    v8h lo = *(const v8h*)(xr + kt * 32);        // K = kt*32 + khalf*8 + 0..7
    v8h hi = *(const v8h*)(xr + kt * 32 + 16);   // K = kt*32 + 16 + khalf*8 + 0..7
    v16h av;
#pragma unroll
    for (int t = 0; t < 8; ++t) { av[t] = lo[t]; av[t + 8] = hi[t]; }
    a[kt] = av;
  }

  // ---- GEMM1: [16 x 160] x [160 x 64] -> f32 [16 x 64] ----
  v8f acc[NNT];
#pragma unroll
  for (int nt = 0; nt < NNT; ++nt) {
    v8f c = {};
#pragma unroll
    for (int kt = 0; kt < NKT1; ++kt) {
      const _Float16* bp = sW1 + (kt * NNT + nt) * 512 + lane * 16;
      v8h blo = *(const v8h*)bp;
      v8h bhi = *(const v8h*)(bp + 8);
      v16h bf;
#pragma unroll
      for (int t = 0; t < 8; ++t) { bf[t] = blo[t]; bf[t + 8] = bhi[t]; }
      c = __builtin_amdgcn_wmma_f32_16x16x32_f16(false, a[kt], false, bf,
                                                 (short)0, c, false, false);
    }
    acc[nt] = c;
  }

  // ---- bias + relu, restage as f16 row-major [16 x 64] (C-layout -> memory) ----
  // NOTE: sH1 region is wave-private; LDS ops of one wave are kept in order
  // (DScnt-tracked), so no workgroup barrier is needed between the stores
  // below and the A-fragment reloads that follow. Each wave flows straight
  // into GEMM2 without waiting for its siblings.
  _Float16* hbase = sH1 + wave * (16 * H_SZ);
#pragma unroll
  for (int nt = 0; nt < NNT; ++nt) {
    const int h = nt * 16 + lrow;
    const float bb = b1[n * H_SZ + h];
#pragma unroll
    for (int i = 0; i < 8; ++i) {
      float f = acc[nt][i] + bb;                 // row = khalf*8 + i, col = h
      f = f > 0.f ? f : 0.f;
      hbase[(khalf * 8 + i) * H_SZ + h] = (_Float16)f;
    }
  }

  // ---- A fragments of h1 from LDS (same-wave data, in-order LDS) ----
  v16h a2[NKT2];
  const _Float16* hr = hbase + lrow * H_SZ + khalf * 8;
#pragma unroll
  for (int kt = 0; kt < NKT2; ++kt) {
    v8h lo = *(const v8h*)(hr + kt * 32);
    v8h hi = *(const v8h*)(hr + kt * 32 + 16);
    v16h av;
#pragma unroll
    for (int t = 0; t < 8; ++t) { av[t] = lo[t]; av[t + 8] = hi[t]; }
    a2[kt] = av;
  }

  // ---- GEMM2: [16 x 64] x [64 x 64] -> f32 [16 x 64] ----
  v8f acc2[NNT];
#pragma unroll
  for (int nt = 0; nt < NNT; ++nt) {
    v8f c = {};
#pragma unroll
    for (int kt = 0; kt < NKT2; ++kt) {
      const _Float16* bp = sW2 + (kt * NNT + nt) * 512 + lane * 16;
      v8h blo = *(const v8h*)bp;
      v8h bhi = *(const v8h*)(bp + 8);
      v16h bf;
#pragma unroll
      for (int t = 0; t < 8; ++t) { bf[t] = blo[t]; bf[t + 8] = bhi[t]; }
      c = __builtin_amdgcn_wmma_f32_16x16x32_f16(false, a2[kt], false, bf,
                                                 (short)0, c, false, false);
    }
    acc2[nt] = c;
  }

  // ---- bias + relu + dot with W3 (per-lane partials over this lane's column) ----
  float part[8] = {0.f, 0.f, 0.f, 0.f, 0.f, 0.f, 0.f, 0.f};
#pragma unroll
  for (int nt = 0; nt < NNT; ++nt) {
    const int h = nt * 16 + lrow;
    const float bb = b2[n * H_SZ + h];
    const float w3 = W3[n * H_SZ + h];
#pragma unroll
    for (int i = 0; i < 8; ++i) {
      float f = acc2[nt][i] + bb;
      f = f > 0.f ? f : 0.f;
      part[i] += f * w3;
    }
  }

  // reduce over the 16 lanes of each half-wave (columns 0..63 all covered)
#pragma unroll
  for (int off = 1; off < 16; off <<= 1) {
#pragma unroll
    for (int i = 0; i < 8; ++i) part[i] += __shfl_xor(part[i], off, 32);
  }

  if (lrow == 0) {
    const float bn = b3[n];
#pragma unroll
    for (int i = 0; i < 8; ++i) {
      const int row = khalf * 8 + i;             // batch row within tile
      out[(size_t)(r0 + row) * D_SZ + n] = part[i] + bn;
    }
  }
}

// ---------------------------------------------------------------------------
// Host side
// ---------------------------------------------------------------------------
extern "C" void kernel_launch(void* const* d_in, const int* in_sizes, int n_in,
                              void* d_out, int out_size, void* d_ws, size_t ws_size,
                              hipStream_t stream) {
  (void)in_sizes; (void)n_in; (void)out_size; (void)ws_size;

  const float* inputs = (const float*)d_in[0];   // [B, D]
  const float* adj    = (const float*)d_in[1];   // [D, D]
  const float* W1     = (const float*)d_in[2];   // [D, D, H]
  const float* b1     = (const float*)d_in[3];   // [D, H]
  const float* W2     = (const float*)d_in[4];   // [D, H, H]
  const float* b2     = (const float*)d_in[5];   // [D, H]
  const float* W3     = (const float*)d_in[6];   // [D, H]
  const float* b3     = (const float*)d_in[7];   // [D]
  const int*   dmask  = (const int*)d_in[8];     // [D] int32
  float* out = (float*)d_out;                    // [B, D] f32

  // Workspace layout (all 16B aligned): ~6.1 MB total
  char* ws = (char*)d_ws;
  _Float16* xpad = (_Float16*)(ws);                               // 8192*160*2   = 2,621,440
  _Float16* w1f  = (_Float16*)(ws + 2621440);                     // 129*10240*2  = 2,641,920
  _Float16* w2f  = (_Float16*)(ws + 2621440 + 2641920);           // 129*4096*2   = 1,056,768

  prep_x_kernel<<<(B_SZ * KPAD) / 256, 256, 0, stream>>>(inputs, dmask, xpad);
  fold_w1_kernel<<<(D_SZ * NKT1 * NNT * 512) / 256, 256, 0, stream>>>(W1, adj, w1f);
  fold_w2_kernel<<<(D_SZ * NKT2 * NNT * 512) / 256, 256, 0, stream>>>(W2, w2f);
  fused_node_mlp_kernel<<<dim3(B_SZ / 128, D_SZ), 256, 0, stream>>>(
      xpad, w1f, w2f, b1, b2, W3, b3, out);
}
